// DeepEMDLoss_377957122735
// MI455X (gfx1250) — matrix operand order repaired and checked by reading, not compiled, not run on hardware
//
#include <hip/hip_runtime.h>
#include <hip/hip_bf16.h>
#include <stdint.h>

#define NB   16
#define CC   512
#define HWW  1024
#define EPS_SINK 0.05f
#define INV_EPS  20.0f
#define SINK_ITERS 50

typedef __attribute__((ext_vector_type(16))) _Float16 v16h;
typedef __attribute__((ext_vector_type(8)))  _Float16 v8h;
typedef __attribute__((ext_vector_type(8)))  float    v8f;
typedef __attribute__((ext_vector_type(4)))  unsigned int v4u;
typedef __attribute__((ext_vector_type(4)))  int      v4i;
typedef __attribute__((ext_vector_type(8)))  int      v8i;

// ---------------- K0: zero f, g, score ----------------
__global__ void k_zero(float* f, float* g, float* score) {
    int i = blockIdx.x * blockDim.x + threadIdx.x;
    if (i < NB * HWW) { f[i] = 0.0f; g[i] = 0.0f; }
    if (i < NB) score[i] = 0.0f;
}

// ---------------- K1: per-(n,c) spatial means of pred & target ----------------
__global__ void k_bmu(const float* __restrict__ pred, const float* __restrict__ target,
                      float* __restrict__ bmu_p, float* __restrict__ bmu_t) {
    __shared__ float sp[256], st[256];
    int nc = blockIdx.x;                       // 0 .. NB*CC-1
    const float* P = pred   + (size_t)nc * HWW;
    const float* T = target + (size_t)nc * HWW;
    float ap = 0.0f, at = 0.0f;
    for (int i = threadIdx.x; i < HWW; i += 256) { ap += P[i]; at += T[i]; }
    sp[threadIdx.x] = ap; st[threadIdx.x] = at; __syncthreads();
    for (int s = 128; s > 0; s >>= 1) {
        if (threadIdx.x < s) { sp[threadIdx.x] += sp[threadIdx.x + s];
                               st[threadIdx.x] += st[threadIdx.x + s]; }
        __syncthreads();
    }
    if (threadIdx.x == 0) {
        bmu_p[nc] = sp[0] * (1.0f / HWW);
        bmu_t[nc] = st[0] * (1.0f / HWW);
    }
}

// ---------------- K1b: y_mu[c] = mean over n of bmu_t ----------------
__global__ void k_ymu(const float* __restrict__ bmu_t, float* __restrict__ ymu) {
    int c = blockIdx.x * blockDim.x + threadIdx.x;
    if (c < CC) {
        float s = 0.0f;
        for (int n = 0; n < NB; ++n) s += bmu_t[n * CC + c];
        ymu[c] = s * (1.0f / NB);
    }
}

// ---------------- K2: raw weights (batched matvec over C) ----------------
__global__ void k_weight(const float* __restrict__ pred, const float* __restrict__ target,
                         const float* __restrict__ bmu_p, const float* __restrict__ bmu_t,
                         float* __restrict__ wp, float* __restrict__ wt) {
    __shared__ float bt[CC], bp[CC];
    int n = blockIdx.x >> 2;
    int p = ((blockIdx.x & 3) << 8) + threadIdx.x;
    for (int c = threadIdx.x; c < CC; c += 256) {
        bt[c] = bmu_t[n * CC + c];
        bp[c] = bmu_p[n * CC + c];
    }
    __syncthreads();
    const float* P = pred   + (size_t)n * CC * HWW + p;
    const float* T = target + (size_t)n * CC * HWW + p;
    float sp = 0.0f, st = 0.0f;
    for (int c = 0; c < CC; ++c) {
        sp = fmaf(P[(size_t)c * HWW], bt[c], sp);
        st = fmaf(T[(size_t)c * HWW], bp[c], st);
    }
    wp[n * HWW + p] = fmaxf(sp, 0.0f) + 1e-4f;
    wt[n * HWW + p] = fmaxf(st, 0.0f) + 1e-4f;
}

// ---------------- K3: normalized log-marginals ----------------
__global__ __launch_bounds__(1024)
void k_logab(const float* __restrict__ wp, const float* __restrict__ wt,
             float* __restrict__ loga, float* __restrict__ logb) {
    __shared__ float sa[1024], sb[1024];
    int n = blockIdx.x, p = threadIdx.x;
    float a = wp[n * HWW + p] + 1e-5f;   // relu is a no-op (already >= 1e-4)
    float b = wt[n * HWW + p] + 1e-5f;
    sa[p] = a; sb[p] = b; __syncthreads();
    for (int s = 512; s > 0; s >>= 1) {
        if (p < s) { sa[p] += sa[p + s]; sb[p] += sb[p + s]; }
        __syncthreads();
    }
    loga[n * HWW + p] = __logf(a * ((float)HWW / sa[0]));
    logb[n * HWW + p] = __logf(b * ((float)HWW / sb[0]));
}

// ---------------- K4a: per-(n,p) channel L2 norms of centered tensors ----------------
__global__ void k_norms(const float* __restrict__ pred, const float* __restrict__ target,
                        const float* __restrict__ ymu,
                        float* __restrict__ nx, float* __restrict__ ny) {
    int n = blockIdx.x >> 2;
    int p = ((blockIdx.x & 3) << 8) + threadIdx.x;
    const float* P = pred   + (size_t)n * CC * HWW + p;
    const float* T = target + (size_t)n * CC * HWW + p;
    float sx = 0.0f, sy = 0.0f;
    for (int c = 0; c < CC; ++c) {
        float mu = ymu[c];
        float xv = P[(size_t)c * HWW] - mu;
        float yv = T[(size_t)c * HWW] - mu;
        sx = fmaf(xv, xv, sx);
        sy = fmaf(yv, yv, sy);
    }
    nx[n * HWW + p] = 1.0f / fmaxf(__builtin_sqrtf(sx), 1e-12f);
    ny[n * HWW + p] = 1.0f / fmaxf(__builtin_sqrtf(sy), 1e-12f);
}

// ---------------- K4b: center, normalize, transpose to (n, p, c), cast f16 ----------------
__global__ void k_transpose(const float* __restrict__ pred, const float* __restrict__ target,
                            const float* __restrict__ ymu,
                            const float* __restrict__ nx, const float* __restrict__ ny,
                            _Float16* __restrict__ xh, _Float16* __restrict__ yh) {
    __shared__ float tx[32][33], ty[32][33];
    int n = blockIdx.z, pB = blockIdx.x << 5, cB = blockIdx.y << 5;
    int ti = threadIdx.x, tj = threadIdx.y;          // (32, 8)
#pragma unroll
    for (int i = 0; i < 4; ++i) {
        int c = cB + tj + i * 8, p = pB + ti;
        float mu = ymu[c];
        size_t idx = ((size_t)n * CC + c) * HWW + p;
        tx[tj + i * 8][ti] = pred[idx]   - mu;
        ty[tj + i * 8][ti] = target[idx] - mu;
    }
    __syncthreads();
#pragma unroll
    for (int i = 0; i < 4; ++i) {
        int p = pB + tj + i * 8, c = cB + ti;
        size_t o = ((size_t)n * HWW + p) * CC + c;
        xh[o] = (_Float16)(tx[ti][tj + i * 8] * nx[n * HWW + p]);
        yh[o] = (_Float16)(ty[ti][tj + i * 8] * ny[n * HWW + p]);
    }
}

// ---------------- K5: WMMA GEMM with TDM-staged B panel in LDS ----------------
// Block = 256 threads = 8 waves; block owns a 256(p) x 32(q) macro-tile.
// One TENSOR_LOAD_TO_LDS stages the shared B panel (32 rows x 512 halfs = 32KB);
// all 8 waves read B fragments from LDS (ds_load_b128), A from global (b128 + prefetch).
__global__ __launch_bounds__(256)
void k_gemm_wmma(const _Float16* __restrict__ xh, const _Float16* __restrict__ yh,
                 float* __restrict__ cosOut) {
    __shared__ _Float16 bpanel[32 * CC];            // 32 KB

    const int b     = blockIdx.x;
    const int n     = b >> 7;                        // 128 blocks per batch
    const int t     = b & 127;
    const int pTile = t >> 5;                        // 0..3   (256-row slabs)
    const int qTile = t & 31;                        // 0..31  (32-col panels)
    const int wav   = threadIdx.x >> 5;              // 0..7
    const int lane  = threadIdx.x & 31;
    const int pBase = pTile * 256 + wav * 32;
    const int qBase = qTile * 32;

    const _Float16* A = xh + (size_t)n * HWW * CC;
    const _Float16* B = yh + (size_t)n * HWW * CC;

    // ---- TDM: stage B[qBase..qBase+31][0..511] (f16) into LDS, wave 0 only ----
    if (wav == 0) {
        unsigned long long ga = (unsigned long long)(uintptr_t)(B + (size_t)qBase * CC);
        unsigned ldsAddr = (unsigned)(unsigned long long)(uintptr_t)bpanel;

        // D# group0: count=1 | lds_addr | global_addr[56:0] | type=2
        v4u g0;
        g0.x = 1u;
        g0.y = ldsAddr;
        g0.z = (unsigned)(ga & 0xFFFFFFFFull);
        g0.w = (unsigned)((ga >> 32) & 0x01FFFFFFull) | (2u << 30);

        // D# group1: data_size=1 (2B); tensor 512x32; tile 512x32; stride0=512
        const unsigned td0 = CC, td1 = 32, tl0 = CC, tl1 = 32;
        const unsigned long long s0 = CC, s1 = (unsigned long long)CC * 32;
        v8i g1;
        g1[0] = (int)(1u << 16);                                   // data_size=1
        g1[1] = (int)((td0 & 0xFFFFu) << 16);                      // tensor_dim0[15:0]
        g1[2] = (int)(((td0 >> 16) & 0xFFFFu) | ((td1 & 0xFFFFu) << 16));
        g1[3] = (int)(((td1 >> 16) & 0xFFFFu) | ((tl0 & 0xFFFFu) << 16)); // tile_dim0
        g1[4] = (int)(tl1 & 0xFFFFu);                              // tile_dim1, tile_dim2=0
        g1[5] = (int)(unsigned)(s0 & 0xFFFFFFFFull);               // stride0[31:0]
        g1[6] = (int)((unsigned)((s0 >> 32) & 0xFFFFull) |
                      (unsigned)((s1 & 0xFFFFull) << 16));         // stride0 hi | stride1 lo
        g1[7] = (int)(unsigned)((s1 >> 16) & 0xFFFFFFFFull);       // stride1[47:16]

        v4i g2 = {0, 0, 0, 0};
        v4i g3 = {0, 0, 0, 0};
#if __clang_major__ >= 23
        v8i g4 = {0, 0, 0, 0, 0, 0, 0, 0};
        __builtin_amdgcn_tensor_load_to_lds(g0, g1, g2, g3, g4, 0);
#else
        __builtin_amdgcn_tensor_load_to_lds(g0, g1, g2, g3, 0);
#endif
        __builtin_amdgcn_s_wait_tensorcnt(0);
    }
    __syncthreads();

    const int m  = lane & 15;
    const int hi = lane >> 4;

    v8f acc00 = {}, acc01 = {}, acc10 = {}, acc11 = {};

    // A 16x32 f16 lane layout: lane<16 holds M=lane, K = {k..k+7, k+16..k+23};
    //                          lane>=16 holds M=lane-16, K = {k+8..k+15, k+24..k+31}.
    // B 32x16 f16 lane layout: col = lane&15, K same as A ({k+hi*8..+7, k+16+hi*8..+7}).
    const _Float16* a0p = A + (size_t)(pBase + m) * CC + hi * 8;
    const _Float16* a1p = a0p + (size_t)16 * CC;
    const _Float16* bp0 = &bpanel[(size_t)m * CC + hi * 8];         // LDS
    const _Float16* bp1 = bp0 + (size_t)16 * CC;                    // LDS

    for (int k = 0; k < CC; k += 32) {
        __builtin_prefetch(a0p + 32, 0, 1);

        v8h a0l = *(const v8h*)(a0p);
        v8h a0h = *(const v8h*)(a0p + 16);
        v8h a1l = *(const v8h*)(a1p);
        v8h a1h = *(const v8h*)(a1p + 16);
        v16h fa0 = __builtin_shufflevector(a0l, a0h, 0,1,2,3,4,5,6,7,8,9,10,11,12,13,14,15);
        v16h fa1 = __builtin_shufflevector(a1l, a1h, 0,1,2,3,4,5,6,7,8,9,10,11,12,13,14,15);
        v8h b0l = *(const v8h*)(bp0 + k);
        v8h b0h = *(const v8h*)(bp0 + k + 16);
        v8h b1l = *(const v8h*)(bp1 + k);
        v8h b1h = *(const v8h*)(bp1 + k + 16);
        v16h fb0 = __builtin_shufflevector(b0l, b0h, 0,1,2,3,4,5,6,7,8,9,10,11,12,13,14,15);
        v16h fb1 = __builtin_shufflevector(b1l, b1h, 0,1,2,3,4,5,6,7,8,9,10,11,12,13,14,15);

        acc00 = __builtin_amdgcn_wmma_f32_16x16x32_f16(false, fa0, false, fb0,
                                                       (short)0, acc00, false, false);
        acc01 = __builtin_amdgcn_wmma_f32_16x16x32_f16(false, fa0, false, fb1,
                                                       (short)0, acc01, false, false);
        acc10 = __builtin_amdgcn_wmma_f32_16x16x32_f16(false, fa1, false, fb0,
                                                       (short)0, acc10, false, false);
        acc11 = __builtin_amdgcn_wmma_f32_16x16x32_f16(false, fa1, false, fb1,
                                                       (short)0, acc11, false, false);

        a0p += 32; a1p += 32;
    }

    // D 16x16 f32 layout: row = hi*8 + r, col = lane&15
    float* Cn = cosOut + (size_t)n * HWW * HWW;
#pragma unroll
    for (int r = 0; r < 8; ++r) {
        int row0 = pBase + hi * 8 + r;
        int col0 = qBase + m;
        Cn[(size_t)row0 * HWW + col0]              = acc00[r];
        Cn[(size_t)row0 * HWW + col0 + 16]         = acc01[r];
        Cn[(size_t)(row0 + 16) * HWW + col0]       = acc10[r];
        Cn[(size_t)(row0 + 16) * HWW + col0 + 16]  = acc11[r];
    }
}

// ---------------- K6: in-place row transform cos -> similarity_map ----------------
__global__ void k_simmap(float* __restrict__ S) {
    __shared__ float red[256];
    int row = blockIdx.x;                       // n*HWW + p
    float* R = S + (size_t)row * HWW;
    float v[4], w[4];
    float m = -1e30f;
#pragma unroll
    for (int i = 0; i < 4; ++i) { v[i] = R[threadIdx.x + i * 256]; m = fmaxf(m, v[i]); }
    red[threadIdx.x] = m; __syncthreads();
    for (int s = 128; s > 0; s >>= 1) {
        if (threadIdx.x < s) red[threadIdx.x] = fmaxf(red[threadIdx.x], red[threadIdx.x + s]);
        __syncthreads();
    }
    float cmax = red[0]; __syncthreads();
    float dmin = 1.0f - cmax;                   // min over q of dist = 1 - cos
    float inv  = 1.0f / (dmin + 1e-5f);
    float sum = 0.0f;
#pragma unroll
    for (int i = 0; i < 4; ++i) {
        float dt = (1.0f - v[i]) * inv;
        w[i] = __expf((1.0f - dt) * 2.0f);      // 1/TEMPERATURE = 2
        sum += w[i];
    }
    red[threadIdx.x] = sum; __syncthreads();
    for (int s = 128; s > 0; s >>= 1) {
        if (threadIdx.x < s) red[threadIdx.x] += red[threadIdx.x + s];
        __syncthreads();
    }
    float rinv = 1.0f / red[0];
#pragma unroll
    for (int i = 0; i < 4; ++i) R[threadIdx.x + i * 256] = w[i] * rinv;
}

// ---------------- K7: Sinkhorn f-update (row LSE) ----------------
__global__ void k_fupd(const float* __restrict__ S, const float* __restrict__ g,
                       const float* __restrict__ loga, float* __restrict__ f) {
    __shared__ float rm[256], rs[256];
    int row = blockIdx.x;
    int n = row >> 10;
    const float* Sr = S + (size_t)row * HWW;
    const float* gn = g + n * HWW;
    float m = -1e30f, s = 0.0f;
    for (int q = threadIdx.x; q < HWW; q += 256) {
        float val = (gn[q] - (1.0f - Sr[q])) * INV_EPS;
        if (val > m) { s = s * __expf(m - val) + 1.0f; m = val; }
        else         { s += __expf(val - m); }
    }
    rm[threadIdx.x] = m; rs[threadIdx.x] = s; __syncthreads();
    for (int st = 128; st > 0; st >>= 1) {
        if (threadIdx.x < st) {
            float m1 = rm[threadIdx.x], s1 = rs[threadIdx.x];
            float m2 = rm[threadIdx.x + st], s2 = rs[threadIdx.x + st];
            float M = fmaxf(m1, m2);
            rs[threadIdx.x] = s1 * __expf(m1 - M) + s2 * __expf(m2 - M);
            rm[threadIdx.x] = M;
        }
        __syncthreads();
    }
    if (threadIdx.x == 0)
        f[row] = EPS_SINK * (loga[row] - (rm[0] + __logf(rs[0])));
}

// ---------------- K8: Sinkhorn g-update (column LSE, coalesced) ----------------
__global__ void k_gupd(const float* __restrict__ S, const float* __restrict__ f,
                       const float* __restrict__ logb, float* __restrict__ g) {
    __shared__ float rm[4][64], rs[4][64];
    int n  = blockIdx.x >> 4;
    int qB = (blockIdx.x & 15) << 6;
    int qi = threadIdx.x;                       // 0..63
    int ps = threadIdx.y;                       // 0..3
    int q  = qB + qi;
    const float* Sn = S + (size_t)n * HWW * HWW;
    const float* fn = f + n * HWW;
    float m = -1e30f, s = 0.0f;
    for (int p = ps * 256; p < ps * 256 + 256; ++p) {
        float val = (fn[p] - (1.0f - Sn[(size_t)p * HWW + q])) * INV_EPS;
        if (val > m) { s = s * __expf(m - val) + 1.0f; m = val; }
        else         { s += __expf(val - m); }
    }
    rm[ps][qi] = m; rs[ps][qi] = s; __syncthreads();
    if (ps == 0) {
#pragma unroll
        for (int j = 1; j < 4; ++j) {
            float m2 = rm[j][qi], s2 = rs[j][qi];
            float M = fmaxf(m, m2);
            s = s * __expf(m - M) + s2 * __expf(m2 - M);
            m = M;
        }
        g[n * HWW + q] = EPS_SINK * (logb[n * HWW + q] - (m + __logf(s)));
    }
}

// ---------------- K9: fused score = sum(S * exp((f+g-cost)/eps)) ----------------
__global__ void k_score(const float* __restrict__ S, const float* __restrict__ f,
                        const float* __restrict__ g, float* __restrict__ score) {
    __shared__ float red[256];
    int row = blockIdx.x;
    int n = row >> 10;
    const float* Sr = S + (size_t)row * HWW;
    const float* gn = g + n * HWW;
    float fp = f[row];
    float acc = 0.0f;
    for (int q = threadIdx.x; q < HWW; q += 256) {
        float sm = Sr[q];
        acc += sm * __expf((fp + gn[q] - (1.0f - sm)) * INV_EPS);
    }
    red[threadIdx.x] = acc; __syncthreads();
    for (int s = 128; s > 0; s >>= 1) {
        if (threadIdx.x < s) red[threadIdx.x] += red[threadIdx.x + s];
        __syncthreads();
    }
    if (threadIdx.x == 0) atomicAdd(&score[n], red[0]);
}

// ---------------- K10: loss = mean(-log(score + 1e-8)) ----------------
__global__ void k_loss(const float* __restrict__ score, float* __restrict__ out) {
    if (threadIdx.x == 0 && blockIdx.x == 0) {
        float s = 0.0f;
        for (int n = 0; n < NB; ++n) s += -__logf(score[n] + 1e-8f);
        out[0] = s * (1.0f / NB);
    }
}

extern "C" void kernel_launch(void* const* d_in, const int* in_sizes, int n_in,
                              void* d_out, int out_size, void* d_ws, size_t ws_size,
                              hipStream_t stream) {
    const float* pred   = (const float*)d_in[0];
    const float* target = (const float*)d_in[1];
    float* out = (float*)d_out;

    // ---- workspace layout ----
    char* w = (char*)d_ws;
    size_t off = 0;
    auto alloc = [&](size_t bytes) { void* p = w + off; off = (off + bytes + 255) & ~(size_t)255; return p; };
    float*    bmu_p = (float*)alloc(NB * CC * 4);
    float*    bmu_t = (float*)alloc(NB * CC * 4);
    float*    ymu   = (float*)alloc(CC * 4);
    float*    wp    = (float*)alloc(NB * HWW * 4);
    float*    wt    = (float*)alloc(NB * HWW * 4);
    float*    loga  = (float*)alloc(NB * HWW * 4);
    float*    logb  = (float*)alloc(NB * HWW * 4);
    float*    nx    = (float*)alloc(NB * HWW * 4);
    float*    ny    = (float*)alloc(NB * HWW * 4);
    float*    fvec  = (float*)alloc(NB * HWW * 4);
    float*    gvec  = (float*)alloc(NB * HWW * 4);
    float*    score = (float*)alloc(NB * 4);
    _Float16* xh    = (_Float16*)alloc((size_t)NB * HWW * CC * 2);
    _Float16* yh    = (_Float16*)alloc((size_t)NB * HWW * CC * 2);
    float*    Smat  = (float*)alloc((size_t)NB * HWW * HWW * 4);
    (void)ws_size; (void)in_sizes; (void)n_in; (void)out_size;

    k_zero<<<64, 256, 0, stream>>>(fvec, gvec, score);
    k_bmu<<<NB * CC, 256, 0, stream>>>(pred, target, bmu_p, bmu_t);
    k_ymu<<<2, 256, 0, stream>>>(bmu_t, ymu);
    k_weight<<<NB * 4, 256, 0, stream>>>(pred, target, bmu_p, bmu_t, wp, wt);
    k_logab<<<NB, 1024, 0, stream>>>(wp, wt, loga, logb);
    k_norms<<<NB * 4, 256, 0, stream>>>(pred, target, ymu, nx, ny);
    k_transpose<<<dim3(HWW / 32, CC / 32, NB), dim3(32, 8), 0, stream>>>(
        pred, target, ymu, nx, ny, xh, yh);

    // 2048 blocks; each block = 8 waves over a 256x32 macro-tile with a TDM-staged B panel
    k_gemm_wmma<<<NB * 128, 256, 0, stream>>>(xh, yh, Smat);

    k_simmap<<<NB * HWW, 256, 0, stream>>>(Smat);

    for (int it = 0; it < SINK_ITERS; ++it) {
        k_fupd<<<NB * HWW, 256, 0, stream>>>(Smat, gvec, loga, fvec);
        k_gupd<<<NB * 16, dim3(64, 4), 0, stream>>>(Smat, fvec, logb, gvec);
    }

    k_score<<<NB * HWW, 256, 0, stream>>>(Smat, fvec, gvec, score);
    k_loss<<<1, 32, 0, stream>>>(score, out);
}
